// MultiHeadAttention_62096637166169
// MI455X (gfx1250) — compile-verified, matrix operand order checked by reading, not run on hardware
//
#include <hip/hip_runtime.h>
#include <hip/hip_bf16.h>
#include <math.h>

// ---------------------------------------------------------------------------
// MHA forward for MI455X (gfx1250, wave32).
// Pipeline: [f32->bf16 transpose weights] -> [QKV proj GEMM, bf16 WMMA] ->
//           [flash attention, bf16 WMMA] -> [output proj GEMM, bf16 WMMA].
// All matmuls: v_wmma_f32_16x16x32_bf16 (fp32 accumulate).
// Global->LDS traffic for weights / K / Q tiles uses gfx1250 async-LDS DMA
// (global_load_async_to_lds_b128 + s_wait_asynccnt) with double-buffered LDS
// so the DMA overlaps the WMMA stream; only one barrier per K-step.
// Workspace (72 MB): 4x2MB bf16 W^T, 3x16MB Q/K/V bf16 [B,H,S,DH],
// 16MB attn output bf16 [B,S,H*DH].
// ---------------------------------------------------------------------------

typedef __attribute__((ext_vector_type(16))) __bf16 v16bf;
typedef __attribute__((ext_vector_type(8)))  float  v8f;

constexpr int Bc = 4, Sc = 2048, Dc = 1024, Hc = 16, DHc = 64, HD = 1024;

// gfx1250 async global->LDS 16-byte copy; tracked by ASYNCcnt.
__device__ inline void async_ld_b128(uint32_t lds_byte_addr, const void* gptr) {
  asm volatile("global_load_async_to_lds_b128 %0, %1, off"
               :: "v"(lds_byte_addr), "v"(gptr) : "memory");
}
__device__ inline void wait_async0() {
  asm volatile("s_wait_asynccnt 0" ::: "memory");
}
__device__ inline uint32_t lds_addr(const void* p) {
  return (uint32_t)(uintptr_t)p;  // low 32 bits of shared aperture = LDS offset
}

// Load a 16x16x32-bf16 WMMA A-style fragment (also used for B stored as B^T).
// row_ptr points at (row, k0) of a row-major bf16 LDS tile; per the ISA layout,
// element e holds k = (e<8 ? 0 : 16) + half*8 + (e&7).
__device__ inline v16bf load_frag(const __bf16* row_ptr, int half) {
  const __bf16* p = row_ptr + half * 8;
  v16bf a;
#pragma unroll
  for (int e = 0; e < 8; ++e) a[e] = p[e];
#pragma unroll
  for (int e = 0; e < 8; ++e) a[e + 8] = p[e + 16];
  return a;
}

// Copy a contiguous 16KB bf16 tile (8192 elems) global->LDS via async DMA.
__device__ inline void async_tile_16k(const __bf16* gsrc, __bf16* ldst, int t) {
#pragma unroll
  for (int j = 0; j < 4; ++j) {
    int c = t + j * 256;  // 1024 chunks of 16B
    async_ld_b128(lds_addr(ldst + c * 8), gsrc + c * 8);
  }
}

// Issue async copy of a 128(n) x 32(k) bf16 tile from W^T[N][K] into LDS
// laid out [n][32] (B^T layout for B-fragments).
__device__ inline void issue_b_tile(const __bf16* WbT, int Kd, int n0, int kk,
                                    __bf16* ldst, int t) {
#pragma unroll
  for (int j = 0; j < 2; ++j) {
    int c = t + j * 256;  // 512 chunks: row = c/4, 16B segment = c%4
    int row = c >> 2, seg = c & 3;
    async_ld_b128(lds_addr(ldst + row * 32 + seg * 8),
                  WbT + (size_t)(n0 + row) * Kd + kk + seg * 8);
  }
}

// ---------------------------------------------------------------------------
// One-time: convert fp32 W[K][N] -> bf16 W^T[N][K] (64x64 LDS tile transpose).
__global__ __launch_bounds__(256) void convert_transpose_kernel(
    const float* __restrict__ in, __bf16* __restrict__ out, int K, int N) {
  __shared__ __bf16 tile[64 * 65];
  const int bk = blockIdx.y * 64, bn = blockIdx.x * 64;
  for (int i = threadIdx.x; i < 64 * 64; i += 256) {
    int r = i >> 6, c = i & 63;
    tile[r * 65 + c] = (__bf16)in[(size_t)(bk + r) * N + bn + c];
  }
  __syncthreads();
  for (int i = threadIdx.x; i < 64 * 64; i += 256) {
    int r = i >> 6, c = i & 63;  // r: n-local, c: k-local
    out[(size_t)(bn + r) * K + bk + c] = tile[c * 65 + r];
  }
}

// ---------------------------------------------------------------------------
// 128x128 block GEMM: C = A[M,Kd] * W[Kd,Nn] + bias, W given as W^T[Nn][Kd].
// 8 waves in a 4x2 grid, each owning a 32x64 sub-tile (2x4 WMMA frags).
// Double-buffered LDS; B tiles arrive via async DMA, A tiles are register-
// staged (fp32->bf16 conversion) and stored after the WMMAs of the current
// step so global latency hides under compute.
// MODE 0: A fp32, output bf16 scattered to [B,H,S,DH]   (QKV projections)
// MODE 1: A bf16, output fp32 row-major [M,Nn]          (final projection)
template <typename AType, int MODE>
__global__ __launch_bounds__(256) void gemm128x128(
    const AType* __restrict__ A, const __bf16* __restrict__ WbT,
    const float* __restrict__ bias, void* __restrict__ Cout,
    int M, int Kd, int Nn) {
  __shared__ __attribute__((aligned(16))) __bf16 a_lds[2][128 * 32];
  __shared__ __attribute__((aligned(16))) __bf16 bT_lds[2][128 * 32];

  const int t = threadIdx.x;
  const int m0 = blockIdx.y * 128;
  const int n0 = blockIdx.x * 128;
  const int wid = t >> 5, lane = t & 31;
  const int half = lane >> 4, lanelo = lane & 15;
  const int wm = (wid >> 1) * 32;  // 0,32,64,96
  const int wn = (wid & 1) * 64;   // 0,64

  v8f acc[2][4] = {};
  float astage[16];

  // prologue: stage tile 0
  issue_b_tile(WbT, Kd, n0, 0, bT_lds[0], t);
#pragma unroll
  for (int j = 0; j < 16; ++j) {
    int i = t + j * 256;
    astage[j] = (float)A[(size_t)(m0 + (i >> 5)) * Kd + (i & 31)];
  }
#pragma unroll
  for (int j = 0; j < 16; ++j) a_lds[0][t + j * 256] = (__bf16)astage[j];

  int buf = 0;
  const int niter = Kd >> 5;
  for (int it = 0; it < niter; ++it) {
    wait_async0();
    __syncthreads();

    const int nk = (it + 1) << 5;
    const bool pf = nk < Kd;
    if (pf) {
      issue_b_tile(WbT, Kd, n0, nk, bT_lds[buf ^ 1], t);
#pragma unroll
      for (int j = 0; j < 16; ++j) {
        int i = t + j * 256;
        astage[j] = (float)A[(size_t)(m0 + (i >> 5)) * Kd + nk + (i & 31)];
      }
    }

    v16bf af[2], bfr[4];
#pragma unroll
    for (int fm = 0; fm < 2; ++fm)
      af[fm] = load_frag(&a_lds[buf][(wm + fm * 16 + lanelo) * 32], half);
#pragma unroll
    for (int fn = 0; fn < 4; ++fn)
      bfr[fn] = load_frag(&bT_lds[buf][(wn + fn * 16 + lanelo) * 32], half);
#pragma unroll
    for (int fm = 0; fm < 2; ++fm)
#pragma unroll
      for (int fn = 0; fn < 4; ++fn)
        acc[fm][fn] = __builtin_amdgcn_wmma_f32_16x16x32_bf16(
            false, af[fm], false, bfr[fn], (short)0, acc[fm][fn], false, false);

    if (pf) {
#pragma unroll
      for (int j = 0; j < 16; ++j)
        a_lds[buf ^ 1][t + j * 256] = (__bf16)astage[j];
    }
    buf ^= 1;
  }

#pragma unroll
  for (int fm = 0; fm < 2; ++fm)
#pragma unroll
    for (int fn = 0; fn < 4; ++fn)
#pragma unroll
      for (int r = 0; r < 8; ++r) {
        int gm = m0 + wm + fm * 16 + half * 8 + r;
        int gn = n0 + wn + fn * 16 + lanelo;
        float v = acc[fm][fn][r] + bias[gn];
        if (MODE == 0) {
          int bb = gm >> 11, s = gm & (Sc - 1);  // S = 2048
          int h = gn >> 6, d = gn & 63;
          ((__bf16*)Cout)[(((size_t)bb * Hc + h) * Sc + s) * DHc + d] =
              (__bf16)v;
        } else {
          ((float*)Cout)[(size_t)gm * Nn + gn] = v;
        }
      }
}

// ---------------------------------------------------------------------------
// Flash attention: one workgroup per (128-row q-tile, head, batch).
// 8 waves; wave w owns query rows [w*16, w*16+16). Softmax row stats live in
// a 16-lane half-wave (shfl_xor {1,2,4,8}). Q and K tiles stream in via async
// DMA (double-buffered K); V is register-staged and stored transposed while
// WMMAs run. P converts C-layout -> A-layout through a private LDS strip.
__global__ __launch_bounds__(256) void flash_attn_kernel(
    const __bf16* __restrict__ Q, const __bf16* __restrict__ Kx,
    const __bf16* __restrict__ V, const int* __restrict__ lens,
    __bf16* __restrict__ Oout) {
  __shared__ __attribute__((aligned(16))) __bf16 q_lds[128 * 64];
  __shared__ __attribute__((aligned(16))) __bf16 k_lds[2][128 * 64];  // [kv][d]
  __shared__ __attribute__((aligned(16))) __bf16 vT_lds[2][64 * 128]; // [d][kv]
  __shared__ __attribute__((aligned(16))) __bf16 p_lds[8 * 16 * 128];

  const int t = threadIdx.x;
  const int qt = blockIdx.x, h = blockIdx.y, b = blockIdx.z;
  const int q0 = qt * 128;
  const int wid = t >> 5, lane = t & 31;
  const int half = lane >> 4, lanelo = lane & 15;

  const size_t bh = ((size_t)b * Hc + h) * (size_t)Sc * DHc;
  const __bf16* Qp = Q + bh;
  const __bf16* Kp = Kx + bh;
  const __bf16* Vp = V + bh;

  int L = lens[b];
  if (L < 1) L = 1;
  if (L > Sc) L = Sc;
  const int ntiles = (L + 127) >> 7;

  // prologue: Q and K tile 0 via async DMA; V tile 0 staged + transposed
  async_tile_16k(Qp + (size_t)q0 * 64, q_lds, t);
  async_tile_16k(Kp, k_lds[0], t);
  {
    const uint4* Vg = (const uint4*)Vp;
    uint4 vs[4];
#pragma unroll
    for (int j = 0; j < 4; ++j) vs[j] = Vg[t + j * 256];
#pragma unroll
    for (int j = 0; j < 4; ++j) {
      const __bf16* vb = (const __bf16*)&vs[j];
      int eb = (t + j * 256) * 8;
      int r = eb >> 6, d0 = eb & 63;
#pragma unroll
      for (int w = 0; w < 8; ++w) vT_lds[0][(d0 + w) * 128 + r] = vb[w];
    }
  }

  float row_m[8], row_l[8];
  v8f o[4] = {};
#pragma unroll
  for (int r = 0; r < 8; ++r) { row_m[r] = -1e30f; row_l[r] = 0.f; }

  v16bf aQ[2];
  uint4 vstage[4];
  int buf = 0;

  for (int kt = 0; kt < ntiles; ++kt) {
    wait_async0();
    __syncthreads();
    if (kt == 0) {
#pragma unroll
      for (int kd = 0; kd < 2; ++kd)
        aQ[kd] = load_frag(q_lds + (wid * 16 + lanelo) * 64 + kd * 32, half);
    }

    const bool pf = (kt + 1) < ntiles;
    if (pf) {
      async_tile_16k(Kp + (size_t)(kt + 1) * (128 * 64), k_lds[buf ^ 1], t);
      const uint4* Vg = (const uint4*)(Vp + (size_t)(kt + 1) * (128 * 64));
#pragma unroll
      for (int j = 0; j < 4; ++j) vstage[j] = Vg[t + j * 256];
    }

    // S = Q * K^T  (16 rows x 128 kv cols per wave)
    v8f sf[8];
#pragma unroll
    for (int fn = 0; fn < 8; ++fn) {
      v8f s = {};
#pragma unroll
      for (int kd = 0; kd < 2; ++kd) {
        v16bf bK =
            load_frag(&k_lds[buf][(fn * 16 + lanelo) * 64 + kd * 32], half);
        s = __builtin_amdgcn_wmma_f32_16x16x32_bf16(false, aQ[kd], false, bK,
                                                    (short)0, s, false, false);
      }
      sf[fn] = s;
    }

    // scale + key-padding mask + row max
    float nm[8];
#pragma unroll
    for (int r = 0; r < 8; ++r) nm[r] = row_m[r];
#pragma unroll
    for (int fn = 0; fn < 8; ++fn) {
      int col = kt * 128 + fn * 16 + lanelo;
      float msk = (col < L) ? 0.f : -1e30f;
#pragma unroll
      for (int r = 0; r < 8; ++r) {
        float sv = sf[fn][r] * 0.125f + msk;  // 1/sqrt(64)
        sf[fn][r] = sv;
        nm[r] = fmaxf(nm[r], sv);
      }
    }
#pragma unroll
    for (int r = 0; r < 8; ++r)
#pragma unroll
      for (int md = 1; md <= 8; md <<= 1)
        nm[r] = fmaxf(nm[r], __shfl_xor(nm[r], md, 32));

    float corr[8], rs[8];
#pragma unroll
    for (int r = 0; r < 8; ++r) {
      corr[r] = __expf(row_m[r] - nm[r]);
      row_m[r] = nm[r];
      rs[r] = 0.f;
    }
#pragma unroll
    for (int fn = 0; fn < 8; ++fn)
#pragma unroll
      for (int r = 0; r < 8; ++r) {
        float p = __expf(sf[fn][r] - nm[r]);
        sf[fn][r] = p;
        rs[r] += p;
      }
#pragma unroll
    for (int r = 0; r < 8; ++r) {
#pragma unroll
      for (int md = 1; md <= 8; md <<= 1) rs[r] += __shfl_xor(rs[r], md, 32);
      row_l[r] = row_l[r] * corr[r] + rs[r];
    }
#pragma unroll
    for (int fj = 0; fj < 4; ++fj)
#pragma unroll
      for (int r = 0; r < 8; ++r) o[fj][r] *= corr[r];

    // C-layout P -> per-wave LDS strip (A-layout source)
    __bf16* pw = p_lds + wid * 2048;
#pragma unroll
    for (int fn = 0; fn < 8; ++fn)
#pragma unroll
      for (int r = 0; r < 8; ++r)
        pw[(half * 8 + r) * 128 + fn * 16 + lanelo] = (__bf16)sf[fn][r];

    // O += P * V
#pragma unroll
    for (int kd = 0; kd < 4; ++kd) {
      v16bf aP = load_frag(pw + lanelo * 128 + kd * 32, half);
#pragma unroll
      for (int fj = 0; fj < 4; ++fj) {
        v16bf bV =
            load_frag(&vT_lds[buf][(fj * 16 + lanelo) * 128 + kd * 32], half);
        o[fj] = __builtin_amdgcn_wmma_f32_16x16x32_bf16(
            false, aP, false, bV, (short)0, o[fj], false, false);
      }
    }

    if (pf) {  // store staged V(kt+1) transposed while WMMAs drain
#pragma unroll
      for (int j = 0; j < 4; ++j) {
        const __bf16* vb = (const __bf16*)&vstage[j];
        int eb = (t + j * 256) * 8;
        int r = eb >> 6, d0 = eb & 63;
#pragma unroll
        for (int w = 0; w < 8; ++w)
          vT_lds[buf ^ 1][(d0 + w) * 128 + r] = vb[w];
      }
    }
    buf ^= 1;
  }

  // normalize and write [B,S,H*DH] bf16
  const size_t orow = ((size_t)b * Sc + q0 + wid * 16) * HD + (size_t)h * DHc;
#pragma unroll
  for (int r = 0; r < 8; ++r) {
    float inv = 1.0f / row_l[r];
#pragma unroll
    for (int fj = 0; fj < 4; ++fj)
      Oout[orow + (size_t)(half * 8 + r) * HD + fj * 16 + lanelo] =
          (__bf16)(o[fj][r] * inv);
  }
}

// ---------------------------------------------------------------------------
extern "C" void kernel_launch(void* const* d_in, const int* in_sizes, int n_in,
                              void* d_out, int out_size, void* d_ws,
                              size_t ws_size, hipStream_t stream) {
  (void)in_sizes; (void)n_in; (void)out_size; (void)ws_size;
  const float* xQ = (const float*)d_in[0];
  const float* xK = (const float*)d_in[1];
  const float* xV = (const float*)d_in[2];
  const int* lens = (const int*)d_in[3];
  const float* Wq = (const float*)d_in[4];
  const float* bq = (const float*)d_in[5];
  const float* Wk = (const float*)d_in[6];
  const float* bk = (const float*)d_in[7];
  const float* Wv = (const float*)d_in[8];
  const float* bv = (const float*)d_in[9];
  const float* Wo = (const float*)d_in[10];
  const float* bo = (const float*)d_in[11];

  char* ws = (char*)d_ws;
  __bf16* WqT = (__bf16*)(ws + ((size_t)0 << 20));   // 2 MB each, [N][K]
  __bf16* WkT = (__bf16*)(ws + ((size_t)2 << 20));
  __bf16* WvT = (__bf16*)(ws + ((size_t)4 << 20));
  __bf16* WoT = (__bf16*)(ws + ((size_t)6 << 20));
  __bf16* Qb  = (__bf16*)(ws + ((size_t)8 << 20));   // [B,H,S,DH]
  __bf16* Kb  = (__bf16*)(ws + ((size_t)24 << 20));
  __bf16* Vb  = (__bf16*)(ws + ((size_t)40 << 20));
  __bf16* Ab  = (__bf16*)(ws + ((size_t)56 << 20));  // [B,S,H*DH]

  dim3 blk(256);
  dim3 gt(HD / 64, Dc / 64);  // 16 x 16
  convert_transpose_kernel<<<gt, blk, 0, stream>>>(Wq, WqT, Dc, HD);
  convert_transpose_kernel<<<gt, blk, 0, stream>>>(Wk, WkT, Dc, HD);
  convert_transpose_kernel<<<gt, blk, 0, stream>>>(Wv, WvT, Dc, HD);
  convert_transpose_kernel<<<gt, blk, 0, stream>>>(Wo, WoT, HD, Dc);

  dim3 gp(HD / 128, (Bc * Sc) / 128);  // 8 x 64 tiles
  gemm128x128<float, 0><<<gp, blk, 0, stream>>>(xQ, WqT, bq, Qb, Bc * Sc, Dc, HD);
  gemm128x128<float, 0><<<gp, blk, 0, stream>>>(xK, WkT, bk, Kb, Bc * Sc, Dc, HD);
  gemm128x128<float, 0><<<gp, blk, 0, stream>>>(xV, WvT, bv, Vb, Bc * Sc, Dc, HD);

  dim3 gf(Sc / 128, Hc, Bc);
  flash_attn_kernel<<<gf, blk, 0, stream>>>(Qb, Kb, Vb, lens, Ab);

  dim3 go(Dc / 128, (Bc * Sc) / 128);
  gemm128x128<__bf16, 1><<<go, blk, 0, stream>>>(Ab, WoT, bo, d_out, Bc * Sc,
                                                 HD, Dc);
}
